// HeteroGNN_85658827751634
// MI455X (gfx1250) — matrix operand order repaired and checked by reading, not compiled, not run on hardware
//
#include <hip/hip_runtime.h>
#include <hip/hip_bf16.h>
#include <stdint.h>

typedef __attribute__((ext_vector_type(16))) __bf16          v16bf;
typedef __attribute__((ext_vector_type(16))) unsigned short  v16u;
typedef __attribute__((ext_vector_type(8)))  float           v8f;

#define NEDGE 320000
#define DD 256
#define HH 256
#define NNODE 50000
#define NPAD 50048   // 391 blocks * 128 rows

// hardware bf16 convert (RNE)
__device__ __forceinline__ unsigned short f2bf(float f) {
  union { __bf16 b; unsigned short u; } c;
  c.b = (__bf16)f;
  return c.u;
}
__device__ __forceinline__ float bf2f(unsigned short u) {
  union { unsigned u; float f; } c;
  c.u = (unsigned)u << 16;
  return c.f;
}

union Frag { v16u s; uint4 q[2]; v16bf b; };

__device__ __forceinline__ uint32_t hash_u32(uint32_t x) {
  x ^= x >> 16; x *= 0x7feb352dU; x ^= x >> 15; x *= 0x846ca68bU; x ^= x >> 16;
  return x;
}

// ---- CDNA5 async global->LDS copy (ASYNCcnt-tracked), 16B per lane ----
__device__ __forceinline__ void async_b128(uint32_t lds_byte_addr, const void* gaddr) {
  asm volatile("global_load_async_to_lds_b128 %0, %1, off"
               :: "v"(lds_byte_addr), "v"(gaddr) : "memory");
}
__device__ __forceinline__ void wait_async0() { asm volatile("s_wait_asynccnt 0x0" ::: "memory"); }
__device__ __forceinline__ void wait_async4() { asm volatile("s_wait_asynccnt 0x4" ::: "memory"); }

// ---------------- prep / utility kernels ----------------
__global__ void k_zero(float* __restrict__ p, int n) {
  int i = blockIdx.x * blockDim.x + threadIdx.x;
  if (i < n) p[i] = 0.f;
}

// f32 -> bf16 bulk convert (4 elems/thread)
__global__ void k_cvt_bf16(const float* __restrict__ x, unsigned short* __restrict__ y, int n4) {
  int i = blockIdx.x * blockDim.x + threadIdx.x;
  if (i < n4) {
    float4 v = ((const float4*)x)[i];
    uint2 p;
    p.x = (uint32_t)f2bf(v.x) | ((uint32_t)f2bf(v.y) << 16);
    p.y = (uint32_t)f2bf(v.z) | ((uint32_t)f2bf(v.w) << 16);
    ((uint2*)y)[i] = p;
  }
}

// build transposed, K-concatenated bf16 weights: Wt[col][k], k in [0,512)
__global__ void k_wcat(const float* __restrict__ Wl, const float* __restrict__ Wr,
                       unsigned short* __restrict__ Wt) {
  int gid = blockIdx.x * blockDim.x + threadIdx.x;   // 512*256
  if (gid < 512 * 256) {
    int k = gid >> 8, col = gid & 255;
    float v = (k < 256) ? Wl[k * 256 + col] : Wr[(k - 256) * 256 + col];
    Wt[(size_t)col * 512 + k] = f2bf(v);
  }
}

__global__ void k_count(const int* __restrict__ dst, float* __restrict__ cnt, int E) {
  int e = blockIdx.x * blockDim.x + threadIdx.x;
  if (e < E) atomicAdd(&cnt[dst[e]], 1.0f);
}

// 32 lanes per edge, 16B bf16 gather each -> coalesced 512B row gather; f32 atomics into L2
__global__ void k_scatter(const unsigned short* __restrict__ xbf, const int* __restrict__ src,
                          const int* __restrict__ dst, float* __restrict__ acc, int E) {
  int gid = blockIdx.x * blockDim.x + threadIdx.x;
  int e = gid >> 5;
  int c = gid & 31;
  if (e >= E) return;
  int s = src[e], d = dst[e];
  uint4 raw = ((const uint4*)xbf)[(size_t)s * 32 + c];
  float* a = acc + (size_t)d * DD + c * 8;
  uint32_t w[4] = { raw.x, raw.y, raw.z, raw.w };
  #pragma unroll
  for (int i = 0; i < 4; ++i) {
    atomicAdd(a + 2 * i + 0, bf2f((unsigned short)(w[i] & 0xFFFF)));
    atomicAdd(a + 2 * i + 1, bf2f((unsigned short)(w[i] >> 16)));
  }
}

// mean-normalize f32 accumulators -> bf16 matrix (4 elems/thread)
__global__ void k_norm(const float* __restrict__ acc, const float* __restrict__ cnt,
                       unsigned short* __restrict__ mbf, int rows) {
  int i = blockIdx.x * blockDim.x + threadIdx.x;
  if (i < rows * 64) {
    float4 v = ((const float4*)acc)[i];
    int r = i >> 6;
    float s = 1.f / fmaxf(cnt[r], 1.f);
    uint2 p;
    p.x = (uint32_t)f2bf(v.x * s) | ((uint32_t)f2bf(v.y * s) << 16);
    p.y = (uint32_t)f2bf(v.z * s) | ((uint32_t)f2bf(v.w * s) << 16);
    ((uint2*)mbf)[i] = p;
  }
}

// ---------------- fused dual GEMM: out = dropout(leaky([m|x] @ Wt^T + b)) ----------------
// bf16 everywhere; K=512. Block tile 128x128, 8 waves as 4(M)x2(N), 2x4 WMMA accs per wave.
// Operands staged via async b128 copies into double-buffered LDS (prefetch depth 1).
//   sA: [row][k] stride 40 shorts;  sB: [col][k] stride 40 shorts (weights pre-transposed).
__global__ __launch_bounds__(256) void k_gemm_layer(
    const unsigned short* __restrict__ Am,   // bf16 [*,256]  (K 0..255)
    const unsigned short* __restrict__ Ax,   // bf16 [*,256]  (K 256..511)
    const unsigned short* __restrict__ Wt,   // bf16 [256 cols][512 k]
    const float* __restrict__ bias,
    unsigned short* __restrict__ outbf,      // bf16 [NPAD,256]
    int M, uint32_t dropStream)
{
  __shared__ __align__(16) unsigned short sA[2][128 * 40];
  __shared__ __align__(16) unsigned short sB[2][128 * 40];

  const int tid  = threadIdx.x;
  const int lane = tid & 31;
  const int wid  = tid >> 5;
  const int wm   = wid & 3;
  const int wn   = wid >> 2;
  const int half = lane >> 4;
  const int l16  = lane & 15;

  const int mBase = blockIdx.x * 128;
  const int nBase = blockIdx.y * 128;

  const uint32_t sAb = (uint32_t)(size_t)&sA[0][0];
  const uint32_t sBb = (uint32_t)(size_t)&sB[0][0];
  const uint32_t bufStride = 128 * 40 * 2;   // bytes

  const v8f vzero = {0.f,0.f,0.f,0.f,0.f,0.f,0.f,0.f};
  v8f acc[2][4];
  for (int i = 0; i < 2; ++i) for (int j = 0; j < 4; ++j) acc[i][j] = vzero;

  // 4 async b128 issues/thread/chunk: 2 for A (row,kgroup), 2 for B (col,kgroup)
  auto issue = [&](int kc, int buf) {
    const unsigned short* As = (kc * 32 < 256) ? Am : Ax;
    const int kk = (kc * 32) & 255;
    #pragma unroll
    for (int i = 0; i < 2; ++i) {
      int p = tid + 256 * i;
      int row = p >> 2, kg = p & 3;
      int grow = mBase + row; if (grow >= M) grow = M - 1;
      async_b128(sAb + buf * bufStride + row * 80 + kg * 16,
                 As + (size_t)grow * 256 + kk + kg * 8);
    }
    #pragma unroll
    for (int i = 0; i < 2; ++i) {
      int p = tid + 256 * i;
      int col = p >> 2, kg = p & 3;
      async_b128(sBb + buf * bufStride + col * 80 + kg * 16,
                 Wt + (size_t)(nBase + col) * 512 + kc * 32 + kg * 8);
    }
  };

  issue(0, 0);
  for (int kc = 0; kc < 16; ++kc) {
    const int cur = kc & 1;
    const bool more = (kc + 1 < 16);
    if (more) { issue(kc + 1, cur ^ 1); wait_async4(); } else { wait_async0(); }
    __syncthreads();

    Frag af[2], bf[4];
    #pragma unroll
    for (int mi = 0; mi < 2; ++mi) {
      int row = wm * 32 + mi * 16 + l16;
      af[mi].q[0] = *(const uint4*)&sA[cur][row * 40 + 8 * half];
      af[mi].q[1] = *(const uint4*)&sA[cur][row * 40 + 16 + 8 * half];
    }
    #pragma unroll
    for (int ni = 0; ni < 4; ++ni) {
      int col = wn * 64 + ni * 16 + l16;
      bf[ni].q[0] = *(const uint4*)&sB[cur][col * 40 + 8 * half];
      bf[ni].q[1] = *(const uint4*)&sB[cur][col * 40 + 16 + 8 * half];
    }
    #pragma unroll
    for (int mi = 0; mi < 2; ++mi)
      #pragma unroll
      for (int ni = 0; ni < 4; ++ni)
        acc[mi][ni] = __builtin_amdgcn_wmma_f32_16x16x32_bf16(
            false, af[mi].b, false, bf[ni].b, (short)0, acc[mi][ni], false, false);
    __syncthreads();
  }

  // epilogue: bias + leaky relu + deterministic dropout, unconditional bf16 stores (padded buf)
  const float invKeep = 1.0f / 0.9f;
  #pragma unroll
  for (int mi = 0; mi < 2; ++mi) {
    #pragma unroll
    for (int ni = 0; ni < 4; ++ni) {
      int gcol = nBase + wn * 64 + ni * 16 + l16;
      float bv = bias[gcol];
      #pragma unroll
      for (int v = 0; v < 8; ++v) {
        int grow = mBase + wm * 32 + mi * 16 + v + 8 * half;
        float val = acc[mi][ni][v] + bv;
        val = (val >= 0.f) ? val : 0.01f * val;
        uint32_t h = hash_u32((uint32_t)(grow * 256 + gcol) * 0x9E3779B9u +
                              dropStream * 0x85EBCA6Bu);
        val = ((h & 0xFFFFFFu) >= 1677722u) ? val * invKeep : 0.f;  // p = 0.1
        outbf[(size_t)grow * HH + gcol] = f2bf(val);
      }
    }
  }
}

// ---------------- output projection: out[M,16] = x @ Wout + bout ----------------
__global__ __launch_bounds__(256) void k_proj(
    const unsigned short* __restrict__ xbf, const float* __restrict__ Wout,
    const float* __restrict__ bout, float* __restrict__ out, int M)
{
  __shared__ __align__(16) unsigned short sX[128 * 40];
  __shared__ __align__(16) unsigned short sW[16 * 264];   // [col][k], stride 264

  const int tid = threadIdx.x;
  #pragma unroll
  for (int i = 0; i < 16; ++i) {
    int e = i * 256 + tid;             // e = k*16 + col
    int k = e >> 4, col = e & 15;
    sW[col * 264 + k] = f2bf(Wout[e]);
  }

  const int lane = tid & 31;
  const int wid  = tid >> 5;
  const int half = lane >> 4;
  const int l16  = lane & 15;
  const int mBase = blockIdx.x * 128;
  const uint32_t sXb = (uint32_t)(size_t)&sX[0];

  v8f acc = {0.f,0.f,0.f,0.f,0.f,0.f,0.f,0.f};

  for (int kc = 0; kc < 8; ++kc) {
    #pragma unroll
    for (int i = 0; i < 2; ++i) {
      int p = tid + 256 * i;
      int row = p >> 2, kg = p & 3;
      int grow = mBase + row; if (grow >= M) grow = M - 1;
      async_b128(sXb + row * 80 + kg * 16,
                 xbf + (size_t)grow * 256 + kc * 32 + kg * 8);
    }
    wait_async0();
    __syncthreads();

    Frag af, bf;
    int row = wid * 16 + l16;
    af.q[0] = *(const uint4*)&sX[row * 40 + 8 * half];
    af.q[1] = *(const uint4*)&sX[row * 40 + 16 + 8 * half];
    bf.q[0] = *(const uint4*)&sW[l16 * 264 + kc * 32 + 8 * half];
    bf.q[1] = *(const uint4*)&sW[l16 * 264 + kc * 32 + 16 + 8 * half];

    acc = __builtin_amdgcn_wmma_f32_16x16x32_bf16(false, af.b, false, bf.b,
                                                  (short)0, acc, false, false);
    __syncthreads();
  }

  float bv = bout[l16];
  #pragma unroll
  for (int v = 0; v < 8; ++v) {
    int grow = mBase + wid * 16 + v + 8 * half;
    if (grow < M) out[(size_t)grow * 16 + l16] = acc[v] + bv;
  }
}

// ---------------- host orchestration ----------------
extern "C" void kernel_launch(void* const* d_in, const int* in_sizes, int n_in,
                              void* d_out, int out_size, void* d_ws, size_t ws_size,
                              hipStream_t stream) {
  const float* x_source = (const float*)d_in[0];
  const float* x_target = (const float*)d_in[1];
  const int*   ei_st    = (const int*)d_in[2];
  const int*   ei_ts    = (const int*)d_in[3];
  const float* Wl_st    = (const float*)d_in[4];
  const float* Wr_st    = (const float*)d_in[5];
  const float* b_st     = (const float*)d_in[6];
  const float* Wl_ts    = (const float*)d_in[7];
  const float* Wr_ts    = (const float*)d_in[8];
  const float* b_ts     = (const float*)d_in[9];
  const float* Wout     = (const float*)d_in[10];
  const float* bout     = (const float*)d_in[11];
  float* out = (float*)d_out;

  const int N = NNODE;
  const size_t NF  = (size_t)N * DD;
  const size_t NFP = (size_t)NPAD * DD;
  const size_t WSZ = 512 * 256;

  // f32 section
  float* acc_t = (float*)d_ws;
  float* acc_s = acc_t + NF;
  float* cnt_t = acc_s + NF;
  float* cnt_s = cnt_t + N;
  // bf16 (ushort) section, 16B aligned
  unsigned short* u = (unsigned short*)(cnt_s + N);
  unsigned short* mbf_t = u;            u += NF;
  unsigned short* mbf_s = u;            u += NF;
  unsigned short* xs0   = u;            u += NF;
  unsigned short* xt0   = u;            u += NF;
  unsigned short* xs1   = u;            u += NFP;
  unsigned short* xt1   = u;            u += NFP;
  unsigned short* xs2   = u;            u += NFP;
  unsigned short* xt2   = u;            u += NFP;
  unsigned short* Wst[2] = { u, u + WSZ };           u += 2 * WSZ;
  unsigned short* Wts[2] = { u, u + WSZ };           u += 2 * WSZ;
  unsigned short* xsL[2] = { xs1, xs2 };
  unsigned short* xtL[2] = { xt1, xt2 };

  const int TPB = 256;
  auto cdiv = [](long a, long b) { return (int)((a + b - 1) / b); };

  // one-time converts: inputs -> bf16, weights -> transposed K-concat bf16
  k_cvt_bf16<<<cdiv((long)NF / 4, TPB), TPB, 0, stream>>>(x_source, xs0, (int)(NF / 4));
  k_cvt_bf16<<<cdiv((long)NF / 4, TPB), TPB, 0, stream>>>(x_target, xt0, (int)(NF / 4));
  for (int l = 0; l < 2; ++l) {
    k_wcat<<<cdiv((long)WSZ, TPB), TPB, 0, stream>>>(Wl_st + (size_t)l * DD * HH,
                                                     Wr_st + (size_t)l * DD * HH, Wst[l]);
    k_wcat<<<cdiv((long)WSZ, TPB), TPB, 0, stream>>>(Wl_ts + (size_t)l * DD * HH,
                                                     Wr_ts + (size_t)l * DD * HH, Wts[l]);
  }

  const unsigned short* cur_s = xs0;
  const unsigned short* cur_t = xt0;
  dim3 ggrid(cdiv(N, 128), HH / 128);

  for (int l = 0; l < 2; ++l) {
    k_zero<<<cdiv((long)NF, TPB), TPB, 0, stream>>>(acc_t, (int)NF);
    k_zero<<<cdiv((long)NF, TPB), TPB, 0, stream>>>(acc_s, (int)NF);
    k_zero<<<cdiv(N, TPB), TPB, 0, stream>>>(cnt_t, N);
    k_zero<<<cdiv(N, TPB), TPB, 0, stream>>>(cnt_s, N);

    // relation source->target
    k_count<<<cdiv(NEDGE, TPB), TPB, 0, stream>>>(ei_st + NEDGE, cnt_t, NEDGE);
    k_scatter<<<cdiv((long)NEDGE * 32, TPB), TPB, 0, stream>>>(cur_s, ei_st, ei_st + NEDGE, acc_t, NEDGE);
    k_norm<<<cdiv((long)N * 64, TPB), TPB, 0, stream>>>(acc_t, cnt_t, mbf_t, N);

    // relation target->source
    k_count<<<cdiv(NEDGE, TPB), TPB, 0, stream>>>(ei_ts + NEDGE, cnt_s, NEDGE);
    k_scatter<<<cdiv((long)NEDGE * 32, TPB), TPB, 0, stream>>>(cur_t, ei_ts, ei_ts + NEDGE, acc_s, NEDGE);
    k_norm<<<cdiv((long)N * 64, TPB), TPB, 0, stream>>>(acc_s, cnt_s, mbf_s, N);

    // new_t / new_s
    k_gemm_layer<<<ggrid, 256, 0, stream>>>(mbf_t, cur_t, Wst[l], b_st + (size_t)l * HH,
                                            xtL[l], N, (uint32_t)(2 * l + 1));
    k_gemm_layer<<<ggrid, 256, 0, stream>>>(mbf_s, cur_s, Wts[l], b_ts + (size_t)l * HH,
                                            xsL[l], N, (uint32_t)(2 * l));
    cur_s = xsL[l];
    cur_t = xtL[l];
  }

  k_proj<<<cdiv(N, 128), 256, 0, stream>>>(cur_s, Wout, bout, out, N);
}
